// MultiHeadAttention_66614942761498
// MI455X (gfx1250) — matrix operand order repaired and checked by reading, not compiled, not run on hardware
//
#include <hip/hip_runtime.h>

// ---------------------------------------------------------------------------
// MHA for MI455X (gfx1250, wave32): bf16 WMMA everywhere, fp32 softmax/accum.
// All activations/weights pre-converted to bf16 once; TDM (tensor_load_to_lds)
// stages bf16 weight panels with TENSORcnt double-buffering so the GEMM inner
// loop is pure ds_load_b128 + v_wmma.
// B=4, Q=S=1024, H=1024, NH=16, HD=64.
// ---------------------------------------------------------------------------

typedef __bf16 bf16;
typedef bf16  v16bf __attribute__((ext_vector_type(16)));
typedef bf16  v8bf  __attribute__((ext_vector_type(8)));
typedef bf16  v4bf  __attribute__((ext_vector_type(4)));
typedef float v8f   __attribute__((ext_vector_type(8)));
typedef unsigned int v4u __attribute__((ext_vector_type(4)));
typedef int   v4i   __attribute__((ext_vector_type(4)));
typedef int   v8i   __attribute__((ext_vector_type(8)));

constexpr int B_  = 4;
constexpr int T_  = 1024;   // Q == S == 1024
constexpr int H_  = 1024;
constexpr int NH_ = 16;
constexpr int HD_ = 64;

#define WMMA_BF16(a, b, c) \
    __builtin_amdgcn_wmma_f32_16x16x32_bf16(false, (a), false, (b), (short)0, (c), false, false)

static __device__ inline v8f zero8() {
    v8f z;
#pragma unroll
    for (int r = 0; r < 8; ++r) z[r] = 0.0f;
    return z;
}

// ---------------------------------------------------------------------------
// TDM: DMA a 2D tile (tile_w x rows elements, row stride in elements) from
// global into LDS at byte offset lds_off. D# layout per CDNA5 ISA.
// dsize_code: 0=1B, 1=2B, 2=4B, 3=8B. 6-arg (clang-23/therock) builtin form.
// ---------------------------------------------------------------------------
static __device__ inline void tdm_load_tile(unsigned lds_off, const void* gptr,
                                            unsigned tensor_w, unsigned rows,
                                            unsigned tile_w, unsigned row_stride,
                                            unsigned dsize_code) {
    unsigned long long ga = (unsigned long long)(size_t)gptr;
    v4u g0 = { 1u,                                  // count=1, is_restore=0, gather=0
               lds_off,                             // lds_addr
               (unsigned)(ga & 0xFFFFFFFFu),        // global_addr[31:0]
               (unsigned)((ga >> 32) & 0x01FFFFFFu) // global_addr[56:32]
                   | (2u << 30) };                  // type=2 ("image")
    v8i g1 = { (int)(dsize_code << 16),                                 // data_size
               (int)((tensor_w & 0xFFFFu) << 16),                       // tensor_dim0[15:0]
               (int)(((tensor_w >> 16) & 0xFFFFu) | ((rows & 0xFFFFu) << 16)), // dim0 hi | tensor_dim1 lo
               (int)(((rows >> 16) & 0xFFFFu) | ((tile_w & 0xFFFFu) << 16)),   // dim1 hi | tile_dim0
               (int)(rows & 0xFFFFu),                                   // tile_dim1 (tile_dim2=0)
               (int)row_stride,                                         // tensor_dim0_stride[31:0]
               0, 0 };
    v4i g2 = {0, 0, 0, 0};
    v4i g3 = {0, 0, 0, 0};
    v8i g4 = {0, 0, 0, 0, 0, 0, 0, 0};
    __builtin_amdgcn_tensor_load_to_lds(g0, g1, g2, g3, g4, 0);
}

// A-fragment (16x32 bf16, M=lane&15) from fp32 row-major source (LDS P tile).
// e=0..15 -> K = k0 + laneHalf*16 + e
static __device__ inline v16bf load_a16_f32(const float* row, int k0, int laneHalf) {
    const float4* p0 = (const float4*)(row + k0 + laneHalf * 16);
    const float4* p1 = (const float4*)(row + k0 + laneHalf * 16 + 8);
    float4 x0 = p0[0], x1 = p0[1], y0 = p1[0], y1 = p1[1];
    v16bf a;
    a[0]  = (bf16)x0.x; a[1]  = (bf16)x0.y; a[2]  = (bf16)x0.z; a[3]  = (bf16)x0.w;
    a[4]  = (bf16)x1.x; a[5]  = (bf16)x1.y; a[6]  = (bf16)x1.z; a[7]  = (bf16)x1.w;
    a[8]  = (bf16)y0.x; a[9]  = (bf16)y0.y; a[10] = (bf16)y0.z; a[11] = (bf16)y0.w;
    a[12] = (bf16)y1.x; a[13] = (bf16)y1.y; a[14] = (bf16)y1.z; a[15] = (bf16)y1.w;
    return a;
}

// Same A-fragment pattern from a bf16 row-major source.
static __device__ inline v16bf load_a16_bf16(const bf16* row, int k0, int laneHalf) {
    v8bf lo = *(const v8bf*)(row + k0 + laneHalf * 16);
    v8bf hi = *(const v8bf*)(row + k0 + laneHalf * 16 + 8);
    v16bf a;
#pragma unroll
    for (int i = 0; i < 8; ++i) { a[i] = lo[i]; a[8 + i] = hi[i]; }
    return a;
}

// B-fragment (32x16 bf16, N=lane&15): element e -> K = laneHalf*16 + e.
// p must already point at (row of column n) + k0 + laneHalf*16 (contiguous 16).
static __device__ inline v16bf load_b16_bf16(const bf16* p) {
    return *(const v16bf*)p;
}

static __device__ inline float wred_max(float v) {
#pragma unroll
    for (int off = 16; off > 0; off >>= 1) v = fmaxf(v, __shfl_xor(v, off, 32));
    return v;
}
static __device__ inline float wred_sum(float v) {
#pragma unroll
    for (int off = 16; off > 0; off >>= 1) v += __shfl_xor(v, off, 32);
    return v;
}

// ---------------------------------------------------------------------------
// Kernel 0a: zero the avg_attn output region (atomics accumulate into it).
// ---------------------------------------------------------------------------
__global__ void zero_f32_kernel(float* __restrict__ p, size_t n) {
    size_t i = (size_t)blockIdx.x * blockDim.x + threadIdx.x;
    size_t stride = (size_t)gridDim.x * blockDim.x;
    for (; i < n; i += stride) p[i] = 0.0f;
}

// ---------------------------------------------------------------------------
// Kernel 0b: fp32 -> bf16 streaming conversion, 4 elements per thread-step.
// ---------------------------------------------------------------------------
__global__ void cvt_f32_to_bf16_kernel(const float4* __restrict__ src,
                                       v4bf* __restrict__ dst, size_t n4) {
    size_t i = (size_t)blockIdx.x * blockDim.x + threadIdx.x;
    size_t stride = (size_t)gridDim.x * blockDim.x;
    for (; i < n4; i += stride) {
        float4 x = src[i];
        v4bf y;
        y[0] = (bf16)x.x; y[1] = (bf16)x.y; y[2] = (bf16)x.z; y[3] = (bf16)x.w;
        dst[i] = y;
    }
}

// ---------------------------------------------------------------------------
// Shared GEMM body: one block (8 waves) computes a 128(M)x64(N) tile of
// Y = X @ W^T with bf16 operands. W 64x32 bf16 k-panels are TDM-staged into
// double-buffered LDS (4KB each). Inner loop: ds_load_b128 + v_wmma only.
// ---------------------------------------------------------------------------
static __device__ inline void gemm_body(const bf16* __restrict__ W, int obase,
                                        const bf16* __restrict__ arow,
                                        bf16* wtile /*[2][64*32] LDS*/, v8f acc[4],
                                        int wv, int laneLo, int laneHalf) {
#pragma unroll
    for (int nt = 0; nt < 4; ++nt) acc[nt] = zero8();

    const unsigned lds0 = (unsigned)(size_t)(void*)wtile;
    const unsigned lds1 = lds0 + 64 * 32 * 2;

    if (wv == 0)
        tdm_load_tile(lds0, W + (size_t)obase * H_, H_, 64, 32, H_, 1u);

    int cur = 0;
    for (int k0 = 0; k0 < H_; k0 += 32) {
        if (wv == 0) __builtin_amdgcn_s_wait_tensorcnt(0);
        __syncthreads();                       // W panel [cur] is ready
        if (wv == 0 && (k0 + 32) < H_)         // prefetch next panel
            tdm_load_tile(cur ? lds0 : lds1,
                          W + (size_t)obase * H_ + k0 + 32, H_, 64, 32, H_, 1u);

        v16bf a = load_a16_bf16(arow, k0, laneHalf);
        const bf16* wbuf = wtile + cur * (64 * 32);
#pragma unroll
        for (int nt = 0; nt < 4; ++nt) {
            v16bf bfrag = load_b16_bf16(wbuf + (nt * 16 + laneLo) * 32 + laneHalf * 16);
            acc[nt] = WMMA_BF16(a, bfrag, acc[nt]);
        }
        __syncthreads();                       // done reading [cur] before overwrite
        cur ^= 1;
    }
}

// ---------------------------------------------------------------------------
// Kernel 1: fused QKV projection. Y = Xb @ Wb^T + b, stored bf16 head-split.
// Block = 8 waves = 128(M)x64(N) tile. Jobs: 3 mats * 32 mgroups * 16 n64.
// Q,K -> [B,NH,T,HD]; V -> transposed [B,NH,HD,S].
// ---------------------------------------------------------------------------
__global__ __launch_bounds__(256) void qkv_proj_kernel(
    const bf16* __restrict__ Xqb, const bf16* __restrict__ Xkb, const bf16* __restrict__ Xvb,
    const bf16* __restrict__ Wqb, const float* __restrict__ bq,
    const bf16* __restrict__ Wkb, const float* __restrict__ bk,
    const bf16* __restrict__ Wvb, const float* __restrict__ bv,
    bf16* __restrict__ Qh, bf16* __restrict__ Kh, bf16* __restrict__ Vt)
{
    __shared__ bf16 wtile[2 * 64 * 32];        // 8 KB double-buffered bf16 W panel

    const int wv       = threadIdx.x >> 5;
    const int lane     = threadIdx.x & 31;
    const int laneLo   = lane & 15;
    const int laneHalf = lane >> 4;

    const int mat    = blockIdx.x / (32 * 16);
    const int rem    = blockIdx.x % (32 * 16);
    const int mgroup = rem / 16;
    const int n64    = rem % 16;

    const bf16*  X    = (mat == 0) ? Xqb : (mat == 1) ? Xkb : Xvb;
    const bf16*  W    = (mat == 0) ? Wqb : (mat == 1) ? Wkb : Wvb;
    const float* bias = (mat == 0) ? bq  : (mat == 1) ? bk  : bv;

    const int tbase = (mgroup * 8 + wv) * 16;
    const int obase = n64 * 64;

    const bf16* arow = X + (size_t)(tbase + laneLo) * H_;
    v8f acc[4];
    gemm_body(W, obase, arow, wtile, acc, wv, laneLo, laneHalf);

    const int bidx    = tbase / T_;
    const int posBase = tbase % T_;
#pragma unroll
    for (int nt = 0; nt < 4; ++nt) {
        const int   o   = obase + nt * 16 + laneLo;
        const float bb_ = bias[o];
        const int   h   = o / HD_;
        const int   d   = o % HD_;
#pragma unroll
        for (int r = 0; r < 8; ++r) {
            const int pos = posBase + r + laneHalf * 8;
            const float y = acc[nt][r] + bb_;
            if (mat == 0)
                Qh[(((size_t)bidx * NH_ + h) * T_ + pos) * HD_ + d] = (bf16)y;
            else if (mat == 1)
                Kh[(((size_t)bidx * NH_ + h) * T_ + pos) * HD_ + d] = (bf16)y;
            else
                Vt[(((size_t)bidx * NH_ + h) * HD_ + d) * T_ + pos] = (bf16)y;
        }
    }
}

// ---------------------------------------------------------------------------
// Kernel 2: attention. One wave per (b, h, 16-row q-tile).
// Scores (128 WMMAs) -> 64KB LDS fp32, masked+scaled at store; softmax with
// wave32 shfl reductions; avg_attn += P/NH (f32 atomics); P·V (128 WMMAs).
// ---------------------------------------------------------------------------
__global__ __launch_bounds__(32) void attn_kernel(
    const bf16* __restrict__ Qh, const bf16* __restrict__ Kh, const bf16* __restrict__ Vt,
    const int* __restrict__ kpm,
    bf16* __restrict__ AO, float* __restrict__ avgOut)
{
    __shared__ float sm[16 * 1024];   // 64 KB score/prob tile

    const int qtile    = blockIdx.x;
    const int h        = blockIdx.y;
    const int b        = blockIdx.z;
    const int lane     = threadIdx.x;
    const int laneLo   = lane & 15;
    const int laneHalf = lane >> 4;
    const int qbase    = qtile * 16;
    const float scale  = 0.125f;      // 1 / sqrt(HD)

    const bf16* Qp  = Qh + (((size_t)b * NH_ + h) * T_ + qbase) * HD_;
    const bf16* Kp  = Kh + (((size_t)b * NH_ + h) * T_) * HD_;
    const bf16* Vp  = Vt + (((size_t)b * NH_ + h) * HD_) * T_;
    const int* mrow = kpm + (size_t)b * T_;

    // ---- Phase 1: scores = Q·K^T, masked & scaled, into LDS ----
    const bf16* qrow = Qp + (size_t)laneLo * HD_;
    for (int st = 0; st < 64; ++st) {
        const int sbase = st * 16;
        v8f acc = zero8();
#pragma unroll
        for (int ds = 0; ds < HD_; ds += 32) {
            v16bf a  = load_a16_bf16(qrow, ds, laneHalf);
            v16bf bb = load_b16_bf16(Kp + (size_t)(sbase + laneLo) * HD_ + ds + laneHalf * 16);
            acc = WMMA_BF16(a, bb, acc);
        }
        const int  scol   = sbase + laneLo;
        const bool masked = (mrow[scol] != 0);
#pragma unroll
        for (int r = 0; r < 8; ++r) {
            float v = masked ? -1e9f : acc[r] * scale;
            sm[(r + laneHalf * 8) * 1024 + scol] = v;
        }
    }

    // ---- Phase 2: softmax per row (all 32 lanes cooperate per row) ----
    for (int m = 0; m < 16; ++m) {
        float* rowp = sm + m * 1024;
        float mx = -3.0e38f;
        for (int i = 0; i < 32; ++i) mx = fmaxf(mx, rowp[i * 32 + lane]);
        mx = wred_max(mx);
        float sum = 0.0f;
        for (int i = 0; i < 32; ++i) {
            const int c = i * 32 + lane;
            const float p = __expf(rowp[c] - mx);
            rowp[c] = p;
            sum += p;
        }
        sum = wred_sum(sum);
        const float inv = 1.0f / sum;
        float* avgrow = avgOut + ((size_t)b * T_ + qbase + m) * T_;
        for (int i = 0; i < 32; ++i) {
            const int c = i * 32 + lane;
            const float pn = rowp[c] * inv;
            rowp[c] = pn;
            atomicAdd(&avgrow[c], pn * (1.0f / (float)NH_));
        }
    }

    // ---- Phase 3: O = P·V, store bf16 to [B,T,H] ----
    const float* prow = sm + laneLo * 1024;
#pragma unroll
    for (int dt = 0; dt < 4; ++dt) {
        const int d0 = dt * 16;
        v8f acc = zero8();
        for (int ss = 0; ss < T_; ss += 32) {
            v16bf a  = load_a16_f32(prow, ss, laneHalf);            // P from LDS, fp32->bf16
            v16bf bb = load_b16_bf16(Vp + (size_t)(d0 + laneLo) * T_ + ss + laneHalf * 16);
            acc = WMMA_BF16(a, bb, acc);
        }
#pragma unroll
        for (int r = 0; r < 8; ++r) {
            const int q = qbase + r + laneHalf * 8;
            AO[((size_t)b * T_ + q) * H_ + h * HD_ + d0 + laneLo] = (bf16)acc[r];
        }
    }
}

// ---------------------------------------------------------------------------
// Kernel 3: output projection. context = AO(bf16) @ Wo^T + bo, fp32 out.
// Block = 8 waves = 128(M)x64(N) tile, TDM-staged bf16 Wo panels.
// ---------------------------------------------------------------------------
__global__ __launch_bounds__(256) void out_proj_kernel(
    const bf16* __restrict__ AO, const bf16* __restrict__ Wob, const float* __restrict__ bo,
    float* __restrict__ ctx)
{
    __shared__ bf16 wtile[2 * 64 * 32];        // 8 KB double-buffered bf16 Wo panel

    const int wv       = threadIdx.x >> 5;
    const int lane     = threadIdx.x & 31;
    const int laneLo   = lane & 15;
    const int laneHalf = lane >> 4;

    const int mgroup = blockIdx.x / 16;        // 0..31
    const int n64    = blockIdx.x % 16;        // 0..15
    const int tbase  = (mgroup * 8 + wv) * 16;
    const int obase  = n64 * 64;

    const bf16* arow = AO + (size_t)(tbase + laneLo) * H_;
    v8f acc[4];
    gemm_body(Wob, obase, arow, wtile, acc, wv, laneLo, laneHalf);

#pragma unroll
    for (int nt = 0; nt < 4; ++nt) {
        const int   o   = obase + nt * 16 + laneLo;
        const float bo_ = bo[o];
#pragma unroll
        for (int r = 0; r < 8; ++r) {
            const int t = tbase + r + laneHalf * 8;
            ctx[(size_t)t * H_ + o] = acc[nt][r] + bo_;
        }
    }
}

// ---------------------------------------------------------------------------
extern "C" void kernel_launch(void* const* d_in, const int* in_sizes, int n_in,
                              void* d_out, int out_size, void* d_ws, size_t ws_size,
                              hipStream_t stream) {
    (void)in_sizes; (void)n_in; (void)out_size; (void)ws_size;

    const float* query = (const float*)d_in[0];
    const float* key   = (const float*)d_in[1];
    const float* value = (const float*)d_in[2];
    const int*   kpm   = (const int*)d_in[3];
    const float* Wq = (const float*)d_in[4];  const float* bq = (const float*)d_in[5];
    const float* Wk = (const float*)d_in[6];  const float* bk = (const float*)d_in[7];
    const float* Wv = (const float*)d_in[8];  const float* bv = (const float*)d_in[9];
    const float* Wo = (const float*)d_in[10]; const float* bo = (const float*)d_in[11];

    const size_t ACT = (size_t)B_ * T_ * H_;           // 4M elems
    const size_t WGT = (size_t)H_ * H_;                // 1M elems

    bf16* base = (bf16*)d_ws;
    bf16* Qh  = base;              // [B,NH,T,HD]
    bf16* Kh  = Qh  + ACT;         // [B,NH,T,HD]
    bf16* Vt  = Kh  + ACT;         // [B,NH,HD,S]
    bf16* AO  = Vt  + ACT;         // [B,T,H]
    bf16* Xqb = AO  + ACT;         // bf16 copies of inputs
    bf16* Xkb = Xqb + ACT;
    bf16* Xvb = Xkb + ACT;
    bf16* Wqb = Xvb + ACT;         // bf16 copies of weights
    bf16* Wkb = Wqb + WGT;
    bf16* Wvb = Wkb + WGT;
    bf16* Wob = Wvb + WGT;         // total 32M bf16 = 64 MB workspace

    float* ctx = (float*)d_out;                        // [B,Q,H]
    float* avg = ctx + (size_t)B_ * T_ * H_;           // [B,Q,S]

    // 0) zero avg_attn (atomically accumulated below)
    zero_f32_kernel<<<4096, 256, 0, stream>>>(avg, (size_t)B_ * T_ * T_);

    // 0b) pre-convert activations + weights to bf16 (one streaming pass)
    cvt_f32_to_bf16_kernel<<<2048, 256, 0, stream>>>((const float4*)query, (v4bf*)Xqb, ACT / 4);
    cvt_f32_to_bf16_kernel<<<2048, 256, 0, stream>>>((const float4*)key,   (v4bf*)Xkb, ACT / 4);
    cvt_f32_to_bf16_kernel<<<2048, 256, 0, stream>>>((const float4*)value, (v4bf*)Xvb, ACT / 4);
    cvt_f32_to_bf16_kernel<<<512,  256, 0, stream>>>((const float4*)Wq,    (v4bf*)Wqb, WGT / 4);
    cvt_f32_to_bf16_kernel<<<512,  256, 0, stream>>>((const float4*)Wk,    (v4bf*)Wkb, WGT / 4);
    cvt_f32_to_bf16_kernel<<<512,  256, 0, stream>>>((const float4*)Wv,    (v4bf*)Wvb, WGT / 4);
    cvt_f32_to_bf16_kernel<<<512,  256, 0, stream>>>((const float4*)Wo,    (v4bf*)Wob, WGT / 4);

    // 1) QKV projections: 3 mats * 32 mgroups * 16 n64 = 1536 blocks, 8 waves each
    qkv_proj_kernel<<<1536, 256, 0, stream>>>(Xqb, Xkb, Xvb,
                                              Wqb, bq, Wkb, bk, Wvb, bv,
                                              Qh, Kh, Vt);

    // 2) attention: grid (qtiles=64, heads=16, batch=4), 1 wave/block
    attn_kernel<<<dim3(64, 16, 4), 32, 0, stream>>>(Qh, Kh, Vt, kpm, AO, avg);

    // 3) output projection: 32 mgroups * 16 n64 = 512 blocks, 8 waves each
    out_proj_kernel<<<512, 256, 0, stream>>>(AO, Wob, bo, ctx);
}
